// MainNet_13460427506005
// MI455X (gfx1250) — matrix operand order repaired and checked by reading, not compiled, not run on hardware
//
#include <hip/hip_runtime.h>
#include <hip/hip_bf16.h>

typedef __attribute__((ext_vector_type(16))) _Float16 v16h;
typedef __attribute__((ext_vector_type(8)))  _Float16 v8h;
typedef __attribute__((ext_vector_type(8)))  float    v8f;
typedef __attribute__((ext_vector_type(4)))  unsigned int u32x4;
typedef __attribute__((ext_vector_type(8)))  int      i32x8;
typedef __attribute__((ext_vector_type(4)))  int      i32x4;

// ---------------------------------------------------------------------------
// Tensor Data Mover: DMA a 2D f16 tile [tileRows x 64] (row-major, stride
// strideElems) from global into LDS, inserting 8 halfs of padding after each
// 64-half row (pad_interval = 32 DWORDs -> code 4, pad_amount = 4 DWORDs ->
// code 3) so the LDS layout has row stride 72 halfs. Tracked by TENSORcnt.
// ---------------------------------------------------------------------------
__device__ __forceinline__ void tdm_load_tile64(const _Float16* gsrc,
                                                const _Float16* ldsDst,
                                                int strideElems, int tileRows) {
  unsigned long long ga = (unsigned long long)(size_t)gsrc;
  unsigned lds_addr = (unsigned)(size_t)ldsDst;   // addr[31:0] == LDS offset

  u32x4 g0;
  g0[0] = 1u;                                        // count=1, user mode
  g0[1] = lds_addr;                                  // lds_addr
  g0[2] = (unsigned)(ga & 0xffffffffull);            // global_addr[31:0]
  g0[3] = (unsigned)((ga >> 32) & 0x01ffffffull)     // global_addr[56:32]
        | (2u << 30);                                // type = 2 ("image")

  unsigned td0   = (unsigned)strideElems;            // tensor_dim0 (elements)
  unsigned td1   = (unsigned)tileRows;               // tensor_dim1
  unsigned tile0 = 64u;                              // BK elements per row
  unsigned tile1 = (unsigned)tileRows;
  unsigned long long s0 = (unsigned long long)(unsigned)strideElems;

  i32x8 g1;
  g1[0] = (int)((1u << 16)      // data_size = 1 -> 2 bytes
              | (1u << 20)      // pad_enable
              | (4u << 22)      // pad_interval: 32 DWORDs
              | (3u << 25));    // pad_amount:   4 DWORDs
  g1[1] = (int)((td0 & 0xffffu) << 16);                            // dim0 lo16
  g1[2] = (int)(((td0 >> 16) & 0xffffu) | ((td1 & 0xffffu) << 16));
  g1[3] = (int)(((td1 >> 16) & 0xffffu) | (tile0 << 16));          // tile_dim0
  g1[4] = (int)(tile1 & 0xffffu);                                  // tile_dim1, tile_dim2=0
  g1[5] = (int)(unsigned)(s0 & 0xffffffffull);                     // dim0_stride lo
  g1[6] = (int)(unsigned)((s0 >> 32) & 0xffffull);                 // dim0_stride hi
  g1[7] = 0;

  i32x4 gz = {0, 0, 0, 0};
#if defined(__clang_major__) && (__clang_major__ >= 23)
  i32x8 gz8 = {0, 0, 0, 0, 0, 0, 0, 0};
  __builtin_amdgcn_tensor_load_to_lds(g0, g1, gz, gz, gz8, 0);
#else
  __builtin_amdgcn_tensor_load_to_lds(g0, g1, gz, gz, 0);
#endif
}

// ---------------------------------------------------------------------------
// f32 -> f16 conversion
// ---------------------------------------------------------------------------
__global__ void k_cvt_f32_f16(const float* __restrict__ in,
                              _Float16* __restrict__ out, long n) {
  long i = (long)blockIdx.x * blockDim.x + threadIdx.x;
  if (i < n) out[i] = (_Float16)in[i];
}

// ---------------------------------------------------------------------------
// im2col for the stride-32 32x32 patchify conv:
// out[b][k=c*1024+kh*32+kw][np] , np in [0,256), valid np<196 (14x14 patches)
// ---------------------------------------------------------------------------
__global__ void k_im2col_stem(const float* __restrict__ x,
                              _Float16* __restrict__ out) {
  long idx = (long)blockIdx.x * blockDim.x + threadIdx.x;
  const long total = 16L * 3072 * 256;
  if (idx >= total) return;
  int np = (int)(idx & 255);
  long t = idx >> 8;
  int k = (int)(t % 3072);
  int b = (int)(t / 3072);
  _Float16 v = (_Float16)0.0f;
  if (np < 196) {
    int c  = k >> 10;
    int r  = k & 1023;
    int kh = r >> 5;
    int kw = r & 31;
    int ph = np / 14, pw = np % 14;
    long src = (((long)(b * 3 + c) * 448) + (ph * 32 + kh)) * 448 + (pw * 32 + kw);
    v = (_Float16)x[src];
  }
  out[idx] = v;
}

// ---------------------------------------------------------------------------
// im2col for 3x3 pad-1 conv on 14x14 maps stored with ld=256 (196 valid)
// ---------------------------------------------------------------------------
__global__ void k_im2col_3x3(const _Float16* __restrict__ y,
                             _Float16* __restrict__ out) {
  long idx = (long)blockIdx.x * blockDim.x + threadIdx.x;
  const long total = 16L * 4608 * 256;
  if (idx >= total) return;
  int np = (int)(idx & 255);
  long t = idx >> 8;
  int k = (int)(t % 4608);
  int b = (int)(t / 4608);
  _Float16 v = (_Float16)0.0f;
  if (np < 196) {
    int c  = k / 9;
    int r9 = k % 9;
    int dy = r9 / 3, dx = r9 % 3;
    int oh = np / 14, ow = np % 14;
    int ih = oh + dy - 1, iw = ow + dx - 1;
    if (ih >= 0 && ih < 14 && iw >= 0 && iw < 14)
      v = y[((long)(b * 512 + c) << 8) + (ih * 14 + iw)];
  }
  out[idx] = v;
}

// ---------------------------------------------------------------------------
// Batched tiled WMMA GEMM:  C[m,n] = sum_k A[m,k]*B[k,n]
//   BM=128, BN=64, BK=64; 256 threads = 8 wave32, each wave owns 32x32 of C.
//   Double-buffered LDS; A tile (and B tile when BTRANS) staged by the Tensor
//   Data Mover for the NEXT iteration while WMMAs run on the current one.
//   BTRANS==0 -> B(k,n)=Bg[k*ldb+n]: global loads issued before compute,
//                transpose ds-scatter after compute (latency hidden).
//   BTRANS==1 -> B(k,n)=Bg[n*ldb+k]: TDM staging.
//   EPI: 0 = f32 store | 1 = f16 store | 2 = f16 +bias ReLU |
//        3 = f16 +bias *gamma+beta ReLU   (f16 paths zero pad cols >= nvalid)
// ---------------------------------------------------------------------------
#define BM 128
#define BN 64
#define BK 64
#define BKP 72   // padded LDS row stride (halfs); 144B

template <int BTRANS, int EPI>
__global__ __launch_bounds__(256)
void k_wmma_gemm(const _Float16* __restrict__ A, const _Float16* __restrict__ Bg,
                 void* __restrict__ Cv,
                 int K, int lda, int ldb, int ldc,
                 long sA, long sB, long sC,
                 int nvalid,
                 const float* __restrict__ bias,
                 const float* __restrict__ gamma,
                 const float* __restrict__ beta) {
  __shared__ _Float16 Al[2][BM * BKP];
  __shared__ _Float16 Bl[2][BN * BKP];

  const int tid  = threadIdx.x;
  const int wave = tid >> 5;
  const int lane = tid & 31;
  const int lr   = lane & 15;
  const int laneHi = lane >> 4;
  const int wm = wave & 3;                // 4 m-slabs of 32 rows
  const int wn = wave >> 2;               // 2 n-slabs of 32 cols

  const int m0 = blockIdx.y * BM;
  const int n0 = blockIdx.x * BN;
  const _Float16* Ab = A  + sA * (long)blockIdx.z;
  const _Float16* Bb = Bg + sB * (long)blockIdx.z;

  v8f acc[2][2];
#pragma unroll
  for (int mi = 0; mi < 2; ++mi)
#pragma unroll
    for (int ni = 0; ni < 2; ++ni)
#pragma unroll
      for (int i = 0; i < 8; ++i) acc[mi][ni][i] = 0.0f;

  // manual B staging (BTRANS==0): 64 k-rows x 8 n-chunks of 8 halfs
  // = 512 chunks; each thread owns chunks {tid, tid+256}
  const int bk0 = tid >> 3;
  const int bn0 = (tid & 7) * 8;
  const int bk1 = bk0 + 32;

  // ---- prologue: stage tile 0 into buffer 0 ----
  if (wave == 0) {
    tdm_load_tile64(Ab + (size_t)m0 * lda, Al[0], lda, BM);
    if (BTRANS)
      tdm_load_tile64(Bb + (size_t)n0 * ldb, Bl[0], ldb, BN);
  }
  if (!BTRANS) {
    v8h b0 = *(const v8h*)(Bb + (size_t)bk0 * ldb + n0 + bn0);
    v8h b1 = *(const v8h*)(Bb + (size_t)bk1 * ldb + n0 + bn0);
#pragma unroll
    for (int u = 0; u < 8; ++u) {
      Bl[0][(bn0 + u) * BKP + bk0] = b0[u];
      Bl[0][(bn0 + u) * BKP + bk1] = b1[u];
    }
  }

  int cur = 0;
  for (int k0 = 0; k0 < K; k0 += BK, cur ^= 1) {
    __builtin_amdgcn_s_wait_tensorcnt(0);   // free for non-issuing waves
    __syncthreads();                        // buffer 'cur' ready; other buffer free

    const bool hasNext = (k0 + BK) < K;
    v8h nb0, nb1;
    if (hasNext) {
      if (wave == 0) {
        tdm_load_tile64(Ab + (size_t)m0 * lda + (k0 + BK), Al[cur ^ 1], lda, BM);
        if (BTRANS)
          tdm_load_tile64(Bb + (size_t)n0 * ldb + (k0 + BK), Bl[cur ^ 1], ldb, BN);
      }
      if (!BTRANS) {
        nb0 = *(const v8h*)(Bb + (size_t)(k0 + BK + bk0) * ldb + n0 + bn0);
        nb1 = *(const v8h*)(Bb + (size_t)(k0 + BK + bk1) * ldb + n0 + bn0);
        if (k0 + 2 * BK < K)
          __builtin_prefetch(Bb + (size_t)(k0 + 2 * BK + bk0) * ldb + n0 + bn0, 0, 3);
      }
    }

    // ---- compute on buffer 'cur': 2 K-substeps x (2x2) WMMA ----
#pragma unroll
    for (int ks = 0; ks < BK; ks += 32) {
      v16h afrag[2];
#pragma unroll
      for (int mi = 0; mi < 2; ++mi) {
        const _Float16* p =
            Al[cur] + (wm * 32 + mi * 16 + lr) * BKP + ks + (laneHi ? 8 : 0);
        v8h lo = *(const v8h*)p;
        v8h hi = *(const v8h*)(p + 16);
#pragma unroll
        for (int i = 0; i < 8; ++i) { afrag[mi][i] = lo[i]; afrag[mi][8 + i] = hi[i]; }
      }
#pragma unroll
      for (int ni = 0; ni < 2; ++ni) {
        const _Float16* p =
            Bl[cur] + (wn * 32 + ni * 16 + lr) * BKP + ks + (laneHi ? 8 : 0);
        v8h lo = *(const v8h*)p;
        v8h hi = *(const v8h*)(p + 16);
        v16h bfrag;
#pragma unroll
        for (int i = 0; i < 8; ++i) { bfrag[i] = lo[i]; bfrag[8 + i] = hi[i]; }
#pragma unroll
        for (int mi = 0; mi < 2; ++mi) {
          acc[mi][ni] = __builtin_amdgcn_wmma_f32_16x16x32_f16(
              false, afrag[mi], false, bfrag, (short)0, acc[mi][ni], false, false);
        }
      }
    }

    // ---- scatter next B tile into the other buffer (after compute) ----
    if (hasNext && !BTRANS) {
#pragma unroll
      for (int u = 0; u < 8; ++u) {
        Bl[cur ^ 1][(bn0 + u) * BKP + bk0] = nb0[u];
        Bl[cur ^ 1][(bn0 + u) * BKP + bk1] = nb1[u];
      }
    }
  }

  // ---- epilogue ----
#pragma unroll
  for (int mi = 0; mi < 2; ++mi) {
    const int rbase = m0 + wm * 32 + mi * 16 + (laneHi ? 8 : 0);
#pragma unroll
    for (int ni = 0; ni < 2; ++ni) {
      const int col = n0 + wn * 32 + ni * 16 + lr;
      if (EPI == 0) {
        float* C = (float*)Cv + sC * (long)blockIdx.z;
#pragma unroll
        for (int r = 0; r < 8; ++r)
          C[(size_t)(rbase + r) * ldc + col] = acc[mi][ni][r];
      } else {
        _Float16* C = (_Float16*)Cv + sC * (long)blockIdx.z;
#pragma unroll
        for (int r = 0; r < 8; ++r) {
          const int row = rbase + r;
          float v = acc[mi][ni][r];
          if (EPI >= 2) v += bias[row];
          if (EPI == 3) v = v * gamma[row] + beta[row];
          if (EPI >= 2) v = fmaxf(v, 0.0f);
          if (col >= nvalid) v = 0.0f;
          C[(size_t)row * ldc + col] = (_Float16)v;
        }
      }
    }
  }
}

// ---------------------------------------------------------------------------
// per-row softmax of (-x) over 512 f32 values -> f16
// ---------------------------------------------------------------------------
__global__ __launch_bounds__(256)
void k_softmax_neg(const float* __restrict__ xxt, _Float16* __restrict__ att) {
  const int row = blockIdx.x;
  const float* p = xxt + (size_t)row * 512;
  _Float16* q = att + (size_t)row * 512;
  const int tid = threadIdx.x;
  float v0 = -p[tid], v1 = -p[tid + 256];
  __shared__ float red[256];
  red[tid] = fmaxf(v0, v1);
  __syncthreads();
  for (int s = 128; s; s >>= 1) {
    if (tid < s) red[tid] = fmaxf(red[tid], red[tid + s]);
    __syncthreads();
  }
  const float mx = red[0];
  __syncthreads();
  const float e0 = __expf(v0 - mx), e1 = __expf(v1 - mx);
  red[tid] = e0 + e1;
  __syncthreads();
  for (int s = 128; s; s >>= 1) {
    if (tid < s) red[tid] += red[tid + s];
    __syncthreads();
  }
  const float inv = 1.0f / red[0];
  q[tid]       = (_Float16)(e0 * inv);
  q[tid + 256] = (_Float16)(e1 * inv);
}

// ---------------------------------------------------------------------------
// mean over one row of 256 f16 (pads are zero) * scale -> f32
// ---------------------------------------------------------------------------
__global__ __launch_bounds__(256)
void k_row_mean(const _Float16* __restrict__ in, float* __restrict__ out,
                float scale) {
  const int row = blockIdx.x;
  const int tid = threadIdx.x;
  __shared__ float red[256];
  red[tid] = (float)in[((size_t)row << 8) + tid];
  __syncthreads();
  for (int s = 128; s; s >>= 1) {
    if (tid < s) red[tid] += red[tid + s];
    __syncthreads();
  }
  if (tid == 0) out[row] = red[0] * scale;
}

// ---------------------------------------------------------------------------
// small FC heads
// ---------------------------------------------------------------------------
__global__ void k_head_fc(const float* __restrict__ vin, const float* __restrict__ W,
                          const float* __restrict__ bvec, float* __restrict__ out,
                          int B, int J, int C, int relu) {
  const int idx = blockIdx.x * blockDim.x + threadIdx.x;
  if (idx >= B * J) return;
  const int b = idx / J, j = idx % J;
  const float* v = vin + (size_t)b * C;
  const float* w = W + (size_t)j * C;
  float s = bvec[j];
  for (int c = 0; c < C; ++c) s += v[c] * w[c];
  if (relu) s = fmaxf(s, 0.0f);
  out[idx] = s;
}

__global__ void k_head_final(const float* __restrict__ h, const float* __restrict__ fc2w,
                             const float* __restrict__ fc2b, const float* __restrict__ raw,
                             float* __restrict__ out) {
  const int idx = blockIdx.x * blockDim.x + threadIdx.x;
  if (idx >= 16 * 200) return;
  const int b = idx / 200, j = idx % 200;
  const float* hv = h + (size_t)b * 200;
  const float* w  = fc2w + (size_t)j * 200;
  float s = fc2b[j];
  for (int t = 0; t < 200; ++t) s += hv[t] * w[t];
  out[idx] = 2.0f * raw[idx] + s;
}

// ---------------------------------------------------------------------------
// launch
// ---------------------------------------------------------------------------
extern "C" void kernel_launch(void* const* d_in, const int* in_sizes, int n_in,
                              void* d_out, int out_size, void* d_ws, size_t ws_size,
                              hipStream_t stream) {
  const float* x        = (const float*)d_in[0];
  const float* stem_w   = (const float*)d_in[1];
  const float* stem_b   = (const float*)d_in[2];
  const float* fc_w     = (const float*)d_in[3];
  const float* fc_b     = (const float*)d_in[4];
  const float* reduce_w = (const float*)d_in[5];
  const float* reduce_b = (const float*)d_in[6];
  const float* r_gamma  = (const float*)d_in[7];
  const float* r_beta   = (const float*)d_in[8];
  const float* conv3_w  = (const float*)d_in[9];
  const float* conv3_b  = (const float*)d_in[10];
  const float* c3_gamma = (const float*)d_in[11];
  const float* c3_beta  = (const float*)d_in[12];
  const float* fc1_w    = (const float*)d_in[13];
  const float* fc1_b    = (const float*)d_in[14];
  const float* fc2_w    = (const float*)d_in[15];
  const float* fc2_b    = (const float*)d_in[16];
  float* out = (float*)d_out;

  char* w = (char*)d_ws;
  size_t o = 0;
  _Float16* stemW  = (_Float16*)(w + o); o += 2048ull * 3072 * 2;
  _Float16* Xh     = (_Float16*)(w + o); o += 16ull * 3072 * 256 * 2;
  _Float16* ffeat  = (_Float16*)(w + o); o += 16ull * 2048 * 256 * 2;
  float*    fmean  = (float*)   (w + o); o += 16ull * 2048 * 4;
  float*    rawb   = (float*)   (w + o); o += 3200ull * 4;
  _Float16* redW   = (_Float16*)(w + o); o += 512ull * 2048 * 2;
  _Float16* rfeat  = (_Float16*)(w + o); o += 16ull * 512 * 256 * 2;
  float*    xxt    = (float*)   (w + o); o += 16ull * 512 * 512 * 4;
  _Float16* att    = (_Float16*)(w + o); o += 16ull * 512 * 512 * 2;
  _Float16* ybuf   = (_Float16*)(w + o); o += 16ull * 512 * 256 * 2;
  _Float16* ycol   = (_Float16*)(w + o); o += 16ull * 4608 * 256 * 2;
  _Float16* c3W    = (_Float16*)(w + o); o += 512ull * 4608 * 2;
  _Float16* zmap   = (_Float16*)(w + o); o += 16ull * 512 * 256 * 2;
  float*    zvec   = (float*)   (w + o); o += 16ull * 512 * 4;
  float*    hbuf   = (float*)   (w + o); o += 3200ull * 4;
  (void)ws_size; (void)n_in; (void)in_sizes; (void)out_size;

  // --- weight conversions ---
  { long n = 2048L * 3072;
    k_cvt_f32_f16<<<(n + 255) / 256, 256, 0, stream>>>(stem_w, stemW, n); }
  { long n = 512L * 2048;
    k_cvt_f32_f16<<<(n + 255) / 256, 256, 0, stream>>>(reduce_w, redW, n); }
  { long n = 512L * 4608;
    k_cvt_f32_f16<<<(n + 255) / 256, 256, 0, stream>>>(conv3_w, c3W, n); }

  // --- stem im2col + GEMM (per batch: [2048x3072] x [3072x256]) ---
  k_im2col_stem<<<(16L * 3072 * 256) / 256, 256, 0, stream>>>(x, Xh);
  {
    dim3 g(256 / BN, 2048 / BM, 16);
    k_wmma_gemm<0, 2><<<g, 256, 0, stream>>>(stemW, Xh, ffeat,
        3072, 3072, 256, 256,
        0L, 3072L * 256, 2048L * 256,
        196, stem_b, nullptr, nullptr);
  }

  // --- GAP + backbone fc -> raw_out ---
  k_row_mean<<<16 * 2048, 256, 0, stream>>>(ffeat, fmean, 1.0f / 196.0f);
  k_head_fc<<<(3200 + 255) / 256, 256, 0, stream>>>(fmean, fc_w, fc_b, rawb,
                                                    16, 200, 2048, 0);

  // --- reduce 1x1 conv + BN + ReLU ---
  {
    dim3 g(256 / BN, 512 / BM, 16);
    k_wmma_gemm<0, 3><<<g, 256, 0, stream>>>(redW, ffeat, rfeat,
        2048, 2048, 256, 256,
        0L, 2048L * 256, 512L * 256,
        196, reduce_b, r_gamma, r_beta);
  }

  // --- xxt = x3 * x3^T (both tiles TDM-staged; K pads are zero) ---
  {
    dim3 g(512 / BN, 512 / BM, 16);
    k_wmma_gemm<1, 0><<<g, 256, 0, stream>>>(rfeat, rfeat, xxt,
        256, 256, 256, 512,
        512L * 256, 512L * 256, 512L * 512,
        512, nullptr, nullptr, nullptr);
  }

  // --- att = softmax(-xxt) per row ---
  k_softmax_neg<<<16 * 512, 256, 0, stream>>>(xxt, att);

  // --- y = att * x3 ---
  {
    dim3 g(256 / BN, 512 / BM, 16);
    k_wmma_gemm<0, 1><<<g, 256, 0, stream>>>(att, rfeat, ybuf,
        512, 512, 256, 256,
        512L * 512, 512L * 256, 512L * 256,
        196, nullptr, nullptr, nullptr);
  }

  // --- conv3 (3x3 pad 1) as im2col GEMM + BN + ReLU ---
  k_im2col_3x3<<<(16L * 4608 * 256) / 256, 256, 0, stream>>>(ybuf, ycol);
  {
    dim3 g(256 / BN, 512 / BM, 16);
    k_wmma_gemm<0, 3><<<g, 256, 0, stream>>>(c3W, ycol, zmap,
        4608, 4608, 256, 256,
        0L, 4608L * 256, 512L * 256,
        196, conv3_b, c3_gamma, c3_beta);
  }

  // --- GAP + two-layer head, final combine ---
  k_row_mean<<<16 * 512, 256, 0, stream>>>(zmap, zvec, 1.0f / 196.0f);
  k_head_fc<<<(3200 + 255) / 256, 256, 0, stream>>>(zvec, fc1_w, fc1_b, hbuf,
                                                    16, 200, 512, 1);
  k_head_final<<<(3200 + 255) / 256, 256, 0, stream>>>(hbuf, fc2_w, fc2_b, rawb, out);
}
